// SSIM_38620345926191
// MI455X (gfx1250) — matrix operand order repaired and checked by reading, not compiled, not run on hardware
//
#include <hip/hip_runtime.h>

typedef __attribute__((ext_vector_type(16))) _Float16 v16h;
typedef __attribute__((ext_vector_type(8)))  _Float16 v8h;
typedef __attribute__((ext_vector_type(8)))  float    v8f;

#define IMG_W   512
#define IMG_H   512
#define OUT_W   502            // 512 - 11 + 1
#define NCH     48             // 16 * 3
#define TILES_X 32
#define STRIPS_Y 4             // each wave does 8 vertical tiles (128 rows)
#define NT       8
#define WAVES_PER_BLOCK 8
#define NBLOCKS ((NCH * TILES_X * STRIPS_Y) / WAVES_PER_BLOCK)   // 768

// SWAPX16 ds_swizzle pattern: xor=0x10, or=0, and=0x1f  -> lane ^ 16
#define SWZ_SWAPX16 0x401F

// ---------------------------------------------------------------------------
// Loads
// ---------------------------------------------------------------------------
template<bool GUARD>
__device__ __forceinline__ void load8(const float* __restrict__ rowp, int col,
                                      bool rowOK, float* __restrict__ out) {
    if (!GUARD) {
        const float4* p = (const float4*)(rowp + col);   // col multiple of 8 -> aligned
        float4 a = p[0], b = p[1];
        out[0] = a.x; out[1] = a.y; out[2] = a.z; out[3] = a.w;
        out[4] = b.x; out[5] = b.y; out[6] = b.z; out[7] = b.w;
    } else if (rowOK && (col + 7) < IMG_W) {
        const float4* p = (const float4*)(rowp + col);
        float4 a = p[0], b = p[1];
        out[0] = a.x; out[1] = a.y; out[2] = a.z; out[3] = a.w;
        out[4] = b.x; out[5] = b.y; out[6] = b.z; out[7] = b.w;
    } else {
#pragma unroll
        for (int e = 0; e < 8; ++e)
            out[e] = (rowOK && (col + e) < IMG_W) ? rowp[col + e] : 0.0f;
    }
}

// One 16-row x 32-col input block (A-layout) for X and Y + f16 products.
template<bool GUARD>
__device__ __forceinline__ void load_block(const float* __restrict__ Xc,
                                           const float* __restrict__ Yc,
                                           int row0, int c0, int m, int hi,
                                           v16h& hx, v16h& hy,
                                           v16h& hxx, v16h& hyy, v16h& hxy) {
    int row = row0 + m;
    bool rowOK = !GUARD || (row < IMG_H);
    if (GUARD && row >= IMG_H) row = IMG_H - 1;          // keep pointer valid
    const float* xr = Xc + (size_t)row * IMG_W;
    const float* yr = Yc + (size_t)row * IMG_W;
    float x1[8], x2[8], y1[8], y2[8];
    load8<GUARD>(xr, c0 + hi * 8,      rowOK, x1);   // i=0..7  : K = hi*8 + i
    load8<GUARD>(xr, c0 + 16 + hi * 8, rowOK, x2);   // i=8..15 : K = 16 + hi*8 + (i-8)
    load8<GUARD>(yr, c0 + hi * 8,      rowOK, y1);
    load8<GUARD>(yr, c0 + 16 + hi * 8, rowOK, y2);
#pragma unroll
    for (int i = 0; i < 8; ++i) {
        hx[i] = (_Float16)x1[i]; hx[i + 8] = (_Float16)x2[i];
        hy[i] = (_Float16)y1[i]; hy[i + 8] = (_Float16)y2[i];
    }
    hxx = hx * hx;
    hyy = hy * hy;
    hxy = hx * hy;
}

// ---------------------------------------------------------------------------
// Blur: horizontal f16-WMMA x2, lane^16 swizzle relayout, vertical f32-WMMA
// ---------------------------------------------------------------------------
__device__ __forceinline__ v8h swz16(v8h v) {
    union { v8h h; int i[4]; } u;
    u.h = v;
#pragma unroll
    for (int k = 0; k < 4; ++k)
        u.i[k] = __builtin_amdgcn_ds_swizzle(u.i[k], SWZ_SWAPX16);
    return u.h;
}

__device__ __forceinline__ v8f blur_field(v16h a0, v16h a1, v16h whB, v16h wvA, bool hi) {
    v8h zh = {};
    v8h d0 = __builtin_amdgcn_wmma_f16_16x16x32_f16(false, a0, false, whB, (short)0, zh, false, false);
    v8h d1 = __builtin_amdgcn_wmma_f16_16x16x32_f16(false, a1, false, whB, (short)0, zh, false, false);
    v8h s0 = swz16(d0);
    v8h s1 = swz16(d1);
    v8h lo = hi ? s1 : d0;   // B element j   -> mid row K = 16*hi + j
    v8h hh = hi ? d1 : s0;   // B element j+8 -> mid row K = 16*hi + 8 + j
    v16h bm;
#pragma unroll
    for (int j = 0; j < 8; ++j) { bm[j] = lo[j]; bm[j + 8] = hh[j]; }
    v8f zf = {};
    return __builtin_amdgcn_wmma_f32_16x16x32_f16(false, wvA, false, bm, (short)0, zf, false, false);
}

// ---------------------------------------------------------------------------
// One vertical strip of NT tiles; GUARD=false is the branch-free interior path.
// ---------------------------------------------------------------------------
template<bool GUARD>
__device__ __forceinline__ float strip_sum(const float* __restrict__ Xc,
                                           const float* __restrict__ Yc,
                                           int r_base, int c0, int m, int hi,
                                           v16h whB, v16h wvA) {
    const float C1 = 1.0e-4f, C2 = 9.0e-4f;
    float sum = 0.0f;

    v16h cx, cy, cxx, cyy, cxy;
    load_block<GUARD>(Xc, Yc, r_base, c0, m, hi, cx, cy, cxx, cyy, cxy);

#pragma unroll
    for (int iy = 0; iy < NT; ++iy) {
        v16h nx, ny, nxx, nyy, nxy;
        load_block<GUARD>(Xc, Yc, r_base + 16 * (iy + 1), c0, m, hi,
                          nx, ny, nxx, nyy, nxy);

        v8f mu1 = blur_field(cx,  nx,  whB, wvA, hi);
        v8f mu2 = blur_field(cy,  ny,  whB, wvA, hi);
        v8f exx = blur_field(cxx, nxx, whB, wvA, hi);
        v8f eyy = blur_field(cyy, nyy, whB, wvA, hi);
        v8f exy = blur_field(cxy, nxy, whB, wvA, hi);

        // SSIM map. f32 D layout: VGPR r -> out row r + 8*hi, col m.
        int ox = c0 + m;
#pragma unroll
        for (int r = 0; r < 8; ++r) {
            int oy = r_base + iy * 16 + r + 8 * hi;
            bool ok = !GUARD || (oy < OUT_W && ox < OUT_W);
            float u1 = mu1[r], u2 = mu2[r];
            float m11 = u1 * u1, m22 = u2 * u2, m12 = u1 * u2;
            float s1  = exx[r] - m11;
            float s2  = eyy[r] - m22;
            float s12 = exy[r] - m12;
            float n1 = 2.0f * s12 + C2, d1v = s1 + s2 + C2;
            float n2 = 2.0f * m12 + C1, d2v = m11 + m22 + C1;
            float v = (n1 * n2) / (d1v * d2v);   // single divide per pixel
            sum += ok ? v : 0.0f;
        }

        cx = nx; cy = ny; cxx = nxx; cyy = nyy; cxy = nxy;
    }
    return sum;
}

__global__ void __launch_bounds__(256)
ssim_tiles(const float* __restrict__ X, const float* __restrict__ Y,
           const float* __restrict__ win, float* __restrict__ partial) {
    const int lane = threadIdx.x & 31;
    const int wave = threadIdx.x >> 5;
    const int g    = blockIdx.x * WAVES_PER_BLOCK + wave;   // 0..6143
    const int tx   = g & (TILES_X - 1);
    const int sy   = (g >> 5) & (STRIPS_Y - 1);
    const int ch   = g >> 7;                                // 0..47
    const int m    = lane & 15;
    const int hi   = lane >> 4;

    // Gaussian weights (uniform -> scalar loads), amortized over NT tiles.
    float w[11];
#pragma unroll
    for (int i = 0; i < 11; ++i) w[i] = win[i];

    // Banded weight operands, built once per wave.
    v16h wvA, whB;
#pragma unroll
    for (int i = 0; i < 16; ++i) {
        int Ka = 16 * hi + i;
        int da = Ka - m;
        wvA[i] = (_Float16)((da >= 0 && da < 11) ? w[da] : 0.0f);
        int Kb = (i < 8) ? (hi * 8 + i) : (16 + hi * 8 + (i - 8));
        int db = Kb - m;
        whB[i] = (_Float16)((db >= 0 && db < 11) ? w[db] : 0.0f);
    }

    const int r_base = sy * (NT * 16);
    const int c0 = tx * 16;
    const float* Xc = X + (size_t)ch * IMG_W * IMG_H;
    const float* Yc = Y + (size_t)ch * IMG_W * IMG_H;

    // Uniform branch: interior strips take the branch-free path.
    float sum;
    if (tx == TILES_X - 1 || sy == STRIPS_Y - 1)
        sum = strip_sum<true >(Xc, Yc, r_base, c0, m, hi, whB, wvA);
    else
        sum = strip_sum<false>(Xc, Yc, r_base, c0, m, hi, whB, wvA);

    // Deterministic block reduction -> one partial per block.
    __shared__ float red[256];
    red[threadIdx.x] = sum;
    __syncthreads();
#pragma unroll
    for (int s = 128; s > 0; s >>= 1) {
        if (threadIdx.x < s) red[threadIdx.x] += red[threadIdx.x + s];
        __syncthreads();
    }
    if (threadIdx.x == 0) partial[blockIdx.x] = red[0];
}

__global__ void __launch_bounds__(256)
reduce_final(const float* __restrict__ partial, int nblk, float* __restrict__ out) {
    __shared__ float red[256];
    float s = 0.0f;
    for (int i = threadIdx.x; i < nblk; i += 256) s += partial[i];   // fixed order
    red[threadIdx.x] = s;
    __syncthreads();
#pragma unroll
    for (int k = 128; k > 0; k >>= 1) {
        if (threadIdx.x < k) red[threadIdx.x] += red[threadIdx.x + k];
        __syncthreads();
    }
    if (threadIdx.x == 0)
        out[0] = red[0] * (1.0f / ((float)NCH * (float)OUT_W * (float)OUT_W));
}

extern "C" void kernel_launch(void* const* d_in, const int* in_sizes, int n_in,
                              void* d_out, int out_size, void* d_ws, size_t ws_size,
                              hipStream_t stream) {
    (void)in_sizes; (void)n_in; (void)out_size; (void)ws_size;
    const float* X   = (const float*)d_in[0];
    const float* Y   = (const float*)d_in[1];
    const float* win = (const float*)d_in[2];
    float* out = (float*)d_out;
    float* partial = (float*)d_ws;   // NBLOCKS floats

    ssim_tiles<<<NBLOCKS, 256, 0, stream>>>(X, Y, win, partial);
    reduce_final<<<1, 256, 0, stream>>>(partial, NBLOCKS, out);
}
